// Block_26087631356643
// MI455X (gfx1250) — matrix operand order repaired and checked by reading, not compiled, not run on hardware
//
#include <hip/hip_runtime.h>

typedef __attribute__((ext_vector_type(16))) _Float16 v16h;
typedef __attribute__((ext_vector_type(8)))  float    v8f;

union Frag16 { v16h v; uint4 q[2]; };

// LDS byte-offset of a pointer into a __shared__ object: generic (flat) pointers
// to LDS carry the LDS offset in their low 32 bits (ISA 10.2 aperture mapping).
__device__ inline unsigned lds_off(const void* p) {
  return (unsigned)(uintptr_t)p;
}

// global -> LDS async copy, 16B per lane, tracked by ASYNCcnt
__device__ inline void async_b128(unsigned ldsoff, const void* gptr, int ioff) {
  if (ioff == 0)
    asm volatile("global_load_async_to_lds_b128 %0, %1, off"
                 :: "v"(ldsoff), "v"((unsigned long long)gptr) : "memory");
  else if (ioff == 16)
    asm volatile("global_load_async_to_lds_b128 %0, %1, off offset:16"
                 :: "v"(ldsoff), "v"((unsigned long long)gptr) : "memory");
  else if (ioff == 32)
    asm volatile("global_load_async_to_lds_b128 %0, %1, off offset:32"
                 :: "v"(ldsoff), "v"((unsigned long long)gptr) : "memory");
  else
    asm volatile("global_load_async_to_lds_b128 %0, %1, off offset:48"
                 :: "v"(ldsoff), "v"((unsigned long long)gptr) : "memory");
}

__device__ inline void wait_async0() {
  asm volatile("s_wait_asynccnt 0x0" ::: "memory");
}

// ---------------------------------------------------------------- cvt f32->f16
__global__ __launch_bounds__(256)
void cvt_f32_f16(const float* __restrict__ in, _Float16* __restrict__ out, size_t n) {
  size_t i = (size_t)blockIdx.x * blockDim.x + threadIdx.x;
  const size_t stride = (size_t)gridDim.x * blockDim.x;
  for (; i < n; i += stride) out[i] = (_Float16)in[i];
}

// ---------------------------------------------------------------- LayerNorm
__global__ __launch_bounds__(256)
void ln_kernel(const float* __restrict__ x, const float* __restrict__ g,
               const float* __restrict__ bt, _Float16* __restrict__ out) {
  constexpr int C = 1024;
  __shared__ float s0[256], s1[256];
  const int row = blockIdx.x;
  const int tid = threadIdx.x;
  const float* xr = x + (size_t)row * C;
  float xv[4], sum = 0.f, sq = 0.f;
#pragma unroll
  for (int i = 0; i < 4; ++i) {
    xv[i] = xr[tid + i * 256];
    sum += xv[i];
    sq  += xv[i] * xv[i];
  }
  s0[tid] = sum; s1[tid] = sq;
  __syncthreads();
  for (int s = 128; s > 0; s >>= 1) {
    if (tid < s) { s0[tid] += s0[tid + s]; s1[tid] += s1[tid + s]; }
    __syncthreads();
  }
  const float mu  = s0[0] * (1.f / C);
  const float var = s1[0] * (1.f / C) - mu * mu;
  const float rs  = rsqrtf(var + 1e-5f);
#pragma unroll
  for (int i = 0; i < 4; ++i) {
    const int c = tid + i * 256;
    out[(size_t)row * C + c] = (_Float16)((xv[i] - mu) * rs * g[c] + bt[c]);
  }
}

// ---------------------------------------------------------------- WMMA GEMM
// C[M,N] = A[M,K](f16) * B[K,N](f16) + bias (+res) (+gelu), 128x128x32 tiles.
#define BM 128
#define BN 128
#define BK 32
#define LDS_STR 40  // halfs; 80B rows keep every fragment chunk 16B aligned

template<int ACT, bool HAS_RES, bool OUT32, bool OUT16>
__global__ __launch_bounds__(256)
void gemm_f16(const _Float16* __restrict__ A, const _Float16* __restrict__ Bw,
              const float* __restrict__ bias, const float* __restrict__ res,
              float* __restrict__ out32, _Float16* __restrict__ out16,
              int M, int N, int K) {
  __shared__ _Float16 As[BM][LDS_STR];
  __shared__ _Float16 BsT[BN][LDS_STR];

  const int tid  = threadIdx.x;
  const int lane = tid & 31;
  const int wid  = tid >> 5;
  const int wm   = wid & 1;    // wave row   (2 x 64 rows)
  const int wn   = wid >> 1;   // wave col   (4 x 32 cols)
  const int hid  = lane >> 4;
  const int lrow = lane & 15;
  const int k0a  = hid * 16;   // A-fragment chunk base
  const int k0b  = hid * 16;   // B-fragment chunk base

  const int bm = blockIdx.y * BM;
  const int bn = blockIdx.x * BN;

  v8f acc[4][2];
#pragma unroll
  for (int mt = 0; mt < 4; ++mt)
#pragma unroll
    for (int nt = 0; nt < 2; ++nt)
#pragma unroll
      for (int v = 0; v < 8; ++v) acc[mt][nt][v] = 0.f;

  const int ar  = tid >> 1;
  const int ak  = (tid & 1) * 16;
  const int bk  = tid >> 3;
  const int bnc = (tid & 7) * 16;

  const unsigned asOff = lds_off(&As[ar][ak]);

  for (int kb = 0; kb < K; kb += BK) {
    // stage A via async global->LDS DMA (contiguous both sides)
    const _Float16* ap = A + (size_t)(bm + ar) * K + kb + ak;
    async_b128(asOff, ap, 0);
    async_b128(asOff, ap, 16);

    // stage B transposed (VGPR round-trip needed for the transpose)
    const _Float16* bp = Bw + (size_t)(kb + bk) * N + bn + bnc;
    Frag16 bt;
    bt.q[0] = *(const uint4*)bp;
    bt.q[1] = *(const uint4*)(bp + 8);
#pragma unroll
    for (int i = 0; i < 16; ++i) BsT[bnc + i][bk] = bt.v[i];

    // prefetch next k-step's B rows into L2 while we compute
    if (kb + BK < K)
      __builtin_prefetch(Bw + (size_t)(kb + BK + bk) * N + bn + bnc, 0, 0);

    wait_async0();
    __syncthreads();

    Frag16 af[4], bf[2];
#pragma unroll
    for (int mt = 0; mt < 4; ++mt) {
      const int r = wm * 64 + mt * 16 + lrow;
      af[mt].q[0] = *(const uint4*)&As[r][k0a];
      af[mt].q[1] = *(const uint4*)&As[r][k0a + 8];
    }
#pragma unroll
    for (int nt = 0; nt < 2; ++nt) {
      const int c = wn * 32 + nt * 16 + lrow;
      bf[nt].q[0] = *(const uint4*)&BsT[c][k0b];
      bf[nt].q[1] = *(const uint4*)&BsT[c][k0b + 8];
    }
#pragma unroll
    for (int mt = 0; mt < 4; ++mt)
#pragma unroll
      for (int nt = 0; nt < 2; ++nt)
        acc[mt][nt] = __builtin_amdgcn_wmma_f32_16x16x32_f16(
            false, af[mt].v, false, bf[nt].v, (short)0, acc[mt][nt], false, false);
    __syncthreads();
  }

#pragma unroll
  for (int mt = 0; mt < 4; ++mt) {
#pragma unroll
    for (int nt = 0; nt < 2; ++nt) {
      const int gcol = bn + wn * 32 + nt * 16 + lrow;
      const float bv = bias[gcol];
#pragma unroll
      for (int v = 0; v < 8; ++v) {
        const int grow = bm + wm * 64 + mt * 16 + hid * 8 + v;
        float o = acc[mt][nt][v] + bv;
        if (HAS_RES) o += res[(size_t)grow * N + gcol];
        if (ACT == 1) o = 0.5f * o * (1.0f + erff(o * 0.70710678118654752f));
        if (OUT32) out32[(size_t)grow * N + gcol] = o;
        if (OUT16) out16[(size_t)grow * N + gcol] = (_Float16)o;
      }
    }
  }
}

// ---------------------------------------------------------------- flash attention
// grid (T/64, H, B), 128 threads (4 waves), each wave owns 16 query rows.
__global__ __launch_bounds__(128)
void attn_kernel(const _Float16* __restrict__ qkv, _Float16* __restrict__ yh, int T) {
  __shared__ _Float16 Ks[64][72];      // [key][d]
  __shared__ _Float16 VsT[64][72];     // [d][key]
  __shared__ _Float16 Ps[4][16][72];   // per-wave P in row-major for A-frag reload

  const int tid  = threadIdx.x;
  const int lane = tid & 31;
  const int wid  = tid >> 5;
  const int hid  = lane >> 4;
  const int lrow = lane & 15;
  const int k0a  = hid * 16;
  const int k0b  = hid * 16;

  const int qb = blockIdx.x;
  const int h  = blockIdx.y;
  const int b  = blockIdx.z;
  const size_t rowBase = (size_t)b * T;
  const int qr0 = qb * 64 + wid * 16;
  const float scale = 0.125f;  // 1/sqrt(64)

  // Q fragments (A layout), 16 rows x 64 dims = 2 k-steps
  Frag16 qf[2];
  {
    const _Float16* qp = qkv + (rowBase + qr0 + lrow) * 3072 + h * 64;
#pragma unroll
    for (int f = 0; f < 2; ++f) {
      qf[f].q[0] = *(const uint4*)(qp + f * 32 + k0a);
      qf[f].q[1] = *(const uint4*)(qp + f * 32 + k0a + 8);
    }
  }

  const int key  = tid >> 1;
  const int part = (tid & 1) * 32;
  const unsigned ksOff = lds_off(&Ks[key][part]);

  float m_run[8], l_run[8];
  v8f oacc[4];
#pragma unroll
  for (int v = 0; v < 8; ++v) { m_run[v] = -INFINITY; l_run[v] = 0.f; }
#pragma unroll
  for (int dt = 0; dt < 4; ++dt)
#pragma unroll
    for (int v = 0; v < 8; ++v) oacc[dt][v] = 0.f;

  for (int j = 0; j <= qb; ++j) {
    {  // stage K tile via async DMA, V tile transposed via VGPRs
      const _Float16* kp = qkv + (rowBase + j * 64 + key) * 3072 + 1024 + h * 64 + part;
      async_b128(ksOff, kp, 0);
      async_b128(ksOff, kp, 16);
      async_b128(ksOff, kp, 32);
      async_b128(ksOff, kp, 48);
      const _Float16* vp = kp + 1024;
      Frag16 vt0, vt1;
      vt0.q[0] = *(const uint4*)(vp);
      vt0.q[1] = *(const uint4*)(vp + 8);
      vt1.q[0] = *(const uint4*)(vp + 16);
      vt1.q[1] = *(const uint4*)(vp + 24);
#pragma unroll
      for (int i = 0; i < 16; ++i) VsT[part + i][key] = vt0.v[i];
#pragma unroll
      for (int i = 0; i < 16; ++i) VsT[part + 16 + i][key] = vt1.v[i];
      wait_async0();
    }
    __syncthreads();

    // S = Q * K^T   (B-fragment column n = key row of Ks -> contiguous reads)
    v8f sacc[4];
#pragma unroll
    for (int nt = 0; nt < 4; ++nt) {
      Frag16 kf0, kf1;  // both k-steps live -> loads can overlap the XDL pipe
      kf0.q[0] = *(const uint4*)&Ks[nt * 16 + lrow][k0b];
      kf0.q[1] = *(const uint4*)&Ks[nt * 16 + lrow][k0b + 8];
      kf1.q[0] = *(const uint4*)&Ks[nt * 16 + lrow][32 + k0b];
      kf1.q[1] = *(const uint4*)&Ks[nt * 16 + lrow][32 + k0b + 8];
      v8f z;
#pragma unroll
      for (int v = 0; v < 8; ++v) z[v] = 0.f;
      z = __builtin_amdgcn_wmma_f32_16x16x32_f16(
          false, qf[0].v, false, kf0.v, (short)0, z, false, false);
      sacc[nt] = __builtin_amdgcn_wmma_f32_16x16x32_f16(
          false, qf[1].v, false, kf1.v, (short)0, z, false, false);
    }

    // scale, causal mask, online softmax
    float sv[4][8], rmax[8];
#pragma unroll
    for (int v = 0; v < 8; ++v) rmax[v] = -INFINITY;
#pragma unroll
    for (int nt = 0; nt < 4; ++nt) {
      const int kidx = j * 64 + nt * 16 + lrow;
#pragma unroll
      for (int v = 0; v < 8; ++v) {
        const int qidx = qr0 + hid * 8 + v;
        float s = sacc[nt][v] * scale;
        if (kidx > qidx) s = -INFINITY;
        sv[nt][v] = s;
        rmax[v] = fmaxf(rmax[v], s);
      }
    }
#pragma unroll
    for (int v = 0; v < 8; ++v)
#pragma unroll
      for (int m = 1; m < 16; m <<= 1)
        rmax[v] = fmaxf(rmax[v], __shfl_xor(rmax[v], m, 32));

    float corr[8], rsum[8];
#pragma unroll
    for (int v = 0; v < 8; ++v) {
      const float mn = fmaxf(m_run[v], rmax[v]);
      corr[v] = __expf(m_run[v] - mn);
      m_run[v] = mn;
      rsum[v] = 0.f;
    }
#pragma unroll
    for (int nt = 0; nt < 4; ++nt)
#pragma unroll
      for (int v = 0; v < 8; ++v) {
        const float p = __expf(sv[nt][v] - m_run[v]);
        rsum[v] += p;
        Ps[wid][hid * 8 + v][nt * 16 + lrow] = (_Float16)p;
      }
#pragma unroll
    for (int v = 0; v < 8; ++v) {
#pragma unroll
      for (int m = 1; m < 16; m <<= 1)
        rsum[v] += __shfl_xor(rsum[v], m, 32);
      l_run[v] = l_run[v] * corr[v] + rsum[v];
    }
#pragma unroll
    for (int dt = 0; dt < 4; ++dt)
#pragma unroll
      for (int v = 0; v < 8; ++v) oacc[dt][v] *= corr[v];
    __syncthreads();

    // O += P * V
    Frag16 pf[2];
#pragma unroll
    for (int f = 0; f < 2; ++f) {
      pf[f].q[0] = *(const uint4*)&Ps[wid][lrow][f * 32 + k0a];
      pf[f].q[1] = *(const uint4*)&Ps[wid][lrow][f * 32 + k0a + 8];
    }
#pragma unroll
    for (int dt = 0; dt < 4; ++dt) {
      Frag16 vf0, vf1;
      vf0.q[0] = *(const uint4*)&VsT[dt * 16 + lrow][k0b];
      vf0.q[1] = *(const uint4*)&VsT[dt * 16 + lrow][k0b + 8];
      vf1.q[0] = *(const uint4*)&VsT[dt * 16 + lrow][32 + k0b];
      vf1.q[1] = *(const uint4*)&VsT[dt * 16 + lrow][32 + k0b + 8];
      oacc[dt] = __builtin_amdgcn_wmma_f32_16x16x32_f16(
          false, pf[0].v, false, vf0.v, (short)0, oacc[dt], false, false);
      oacc[dt] = __builtin_amdgcn_wmma_f32_16x16x32_f16(
          false, pf[1].v, false, vf1.v, (short)0, oacc[dt], false, false);
    }
    __syncthreads();
  }

#pragma unroll
  for (int dt = 0; dt < 4; ++dt) {
    const int col = h * 64 + dt * 16 + lrow;
#pragma unroll
    for (int v = 0; v < 8; ++v) {
      const int row = qr0 + hid * 8 + v;
      yh[(rowBase + row) * 1024 + col] = (_Float16)(oacc[dt][v] / l_run[v]);
    }
  }
}

// ---------------------------------------------------------------- launch
extern "C" void kernel_launch(void* const* d_in, const int* in_sizes, int n_in,
                              void* d_out, int out_size, void* d_ws, size_t ws_size,
                              hipStream_t stream) {
  (void)in_sizes; (void)n_in; (void)out_size; (void)ws_size;
  constexpr int Bb = 2, T = 2048, C = 1024, H = 16, M = Bb * T;

  const float* x      = (const float*)d_in[0];
  const float* w_qkv  = (const float*)d_in[1];
  const float* b_qkv  = (const float*)d_in[2];
  const float* w_proj = (const float*)d_in[3];
  const float* b_proj = (const float*)d_in[4];
  const float* ln1_g  = (const float*)d_in[5];
  const float* ln1_b  = (const float*)d_in[6];
  const float* ln2_g  = (const float*)d_in[7];
  const float* ln2_b  = (const float*)d_in[8];
  const float* w1     = (const float*)d_in[9];
  const float* b1     = (const float*)d_in[10];
  const float* w2     = (const float*)d_in[11];
  const float* b2     = (const float*)d_in[12];

  char* ws = (char*)d_ws;
  size_t off = 0;
  auto carve = [&](size_t bytes) -> char* {
    char* p = ws + off;
    off += (bytes + 255) & ~(size_t)255;
    return p;
  };
  _Float16* lnh    = (_Float16*)carve((size_t)M * C * 2);        // 8 MB (reused ln1/ln2)
  _Float16* wqkvh  = (_Float16*)carve((size_t)C * 3 * C * 2);    // 6 MB
  _Float16* wprojh = (_Float16*)carve((size_t)C * C * 2);        // 2 MB
  _Float16* w1h    = (_Float16*)carve((size_t)C * 4 * C * 2);    // 8 MB
  _Float16* w2h    = (_Float16*)carve((size_t)4 * C * C * 2);    // 8 MB
  float*    x1     = (float*)   carve((size_t)M * C * 4);        // 16 MB
  char*     big    = carve((size_t)M * 4 * C * 2);               // 32 MB union
  _Float16* qkvh = (_Float16*)big;                               // 24 MB (dead after attn)
  _Float16* yh   = (_Float16*)(big + (size_t)M * 3 * C * 2);     // 8 MB (dead after proj)
  _Float16* hh   = (_Float16*)big;                               // 32 MB (mlp hidden)

  // weight conversion fp32 -> fp16
  cvt_f32_f16<<<2048, 256, 0, stream>>>(w_qkv,  wqkvh,  (size_t)C * 3 * C);
  cvt_f32_f16<<<2048, 256, 0, stream>>>(w_proj, wprojh, (size_t)C * C);
  cvt_f32_f16<<<2048, 256, 0, stream>>>(w1,     w1h,    (size_t)C * 4 * C);
  cvt_f32_f16<<<2048, 256, 0, stream>>>(w2,     w2h,    (size_t)4 * C * C);

  // x -> ln1 -> qkv
  ln_kernel<<<M, 256, 0, stream>>>(x, ln1_g, ln1_b, lnh);
  gemm_f16<0, false, false, true><<<dim3(3 * C / BN, M / BM), 256, 0, stream>>>(
      lnh, wqkvh, b_qkv, nullptr, nullptr, qkvh, M, 3 * C, C);

  // attention
  attn_kernel<<<dim3(T / 64, H, Bb), 128, 0, stream>>>(qkvh, yh, T);

  // proj + residual -> x1
  gemm_f16<0, true, true, false><<<dim3(C / BN, M / BM), 256, 0, stream>>>(
      yh, wprojh, b_proj, x, x1, nullptr, M, C, C);

  // ln2 -> mlp fc1 (+gelu) -> fc2 + residual -> out
  ln_kernel<<<M, 256, 0, stream>>>(x1, ln2_g, ln2_b, lnh);
  gemm_f16<1, false, false, true><<<dim3(4 * C / BN, M / BM), 256, 0, stream>>>(
      lnh, w1h, b1, nullptr, nullptr, hh, M, 4 * C, C);
  gemm_f16<0, true, true, false><<<dim3(C / BN, M / BM), 256, 0, stream>>>(
      hh, w2h, b2, x1, (float*)d_out, nullptr, M, C, 4 * C);
}
